// SelfAttention_23837068493232
// MI455X (gfx1250) — compile-verified, matrix operand order checked by reading, not run on hardware
//
#include <hip/hip_runtime.h>
#include <hip/hip_bf16.h>
#include <math.h>

// ---------------------------------------------------------------------------
// Self-matching additive attention + bidirectional GRU for MI455X (gfx1250).
//  * GEMMs: v_wmma_f32_16x16x32_f16, operands pre-converted to f16 so the
//    K-loop is pure b128 fragment loads + 4 WMMAs per iteration (2x2 tiles).
//  * Score contraction (tanh inside the sum) / softmax / GRU recurrence: FP32.
//  * Score kernel stages tiles into LDS with gfx1250 async loads when the
//    builtin exists (fallback: float4 copies).
// ---------------------------------------------------------------------------

typedef __attribute__((ext_vector_type(16))) _Float16 v16h;
typedef __attribute__((ext_vector_type(8)))  _Float16 v8h;
typedef __attribute__((ext_vector_type(8)))  float    v8f;
typedef __attribute__((ext_vector_type(4)))  int      i32x4;

#define LSEQ 512
#define DDIM 512
#define HDIM 512

#if defined(__HIP_DEVICE_COMPILE__) && __has_builtin(__builtin_amdgcn_global_load_async_to_lds_b128)
#define ASYNC_LDS 1
typedef __attribute__((address_space(1))) i32x4 g_i32x4;   // global b128 payload
typedef __attribute__((address_space(3))) i32x4 l_i32x4;   // LDS b128 payload
#else
#define ASYNC_LDS 0
#endif

// ---------------------------------------------------------------------------
// Elementwise f32 -> f16 conversion (n multiple of 4).
// ---------------------------------------------------------------------------
__global__ __launch_bounds__(256)
void f32_to_f16_kernel(const float* __restrict__ in, _Float16* __restrict__ out,
                       int n) {
    const int i = (blockIdx.x * blockDim.x + threadIdx.x) * 4;
    if (i >= n) return;
    const float4 f = *(const float4*)(in + i);
    out[i + 0] = (_Float16)f.x;
    out[i + 1] = (_Float16)f.y;
    out[i + 2] = (_Float16)f.z;
    out[i + 3] = (_Float16)f.w;
}

// ---------------------------------------------------------------------------
// WMMA GEMM:  C[m][n] = sum_k A[m][k] * W[n][k]  (+ bias[n])
// A: [M][K] f16, W: [N][K] f16, C: [M][N] f32.
// Block = 256 threads = 8 waves (4 M x 2 N); each wave owns a 32x32 tile
// (2x2 WMMA 16x16 tiles) -> block tile 128(M) x 64(N), 4 WMMAs / K-step.
// Requires M % 128 == 0, N % 64 == 0, K % 32 == 0.
// ---------------------------------------------------------------------------
__global__ __launch_bounds__(256)
void gemm_wmma_f16_kernel(const _Float16* __restrict__ A,
                          const _Float16* __restrict__ W,
                          const float* __restrict__ bias,
                          float* __restrict__ C,
                          int M, int N, int K) {
    const int wave = threadIdx.x >> 5;          // 0..7
    const int lane = threadIdx.x & 31;
    const int half = lane >> 4;                 // 0: lanes 0-15, 1: lanes 16-31
    const int r    = lane & 15;

    const int wy = wave >> 1, wx = wave & 1;    // 4 x 2 waves
    const int m0 = blockIdx.y * 128 + wy * 32;
    const int n0 = blockIdx.x * 64  + wx * 32;

    // 16-bit A-matrix 16x32 lane layout (ISA 7.12.2):
    //   half 0 -> K = kb..kb+7 (elems 0..7), kb+16..kb+23 (elems 8..15)
    //   half 1 -> kb = k0 + 8
    const _Float16* __restrict__ a0p = A + (size_t)(m0 + r)      * K;
    const _Float16* __restrict__ a1p = A + (size_t)(m0 + 16 + r) * K;
    const _Float16* __restrict__ b0p = W + (size_t)(n0 + r)      * K;
    const _Float16* __restrict__ b1p = W + (size_t)(n0 + 16 + r) * K;

    v8f acc00 = {}, acc01 = {}, acc10 = {}, acc11 = {};

    for (int k0 = 0; k0 < K; k0 += 32) {
        const int kb = k0 + half * 8;
        const v16h a0 = __builtin_shufflevector(
            *(const v8h*)(a0p + kb), *(const v8h*)(a0p + kb + 16),
            0, 1, 2, 3, 4, 5, 6, 7, 8, 9, 10, 11, 12, 13, 14, 15);
        const v16h a1 = __builtin_shufflevector(
            *(const v8h*)(a1p + kb), *(const v8h*)(a1p + kb + 16),
            0, 1, 2, 3, 4, 5, 6, 7, 8, 9, 10, 11, 12, 13, 14, 15);
        const v16h b0 = __builtin_shufflevector(
            *(const v8h*)(b0p + kb), *(const v8h*)(b0p + kb + 16),
            0, 1, 2, 3, 4, 5, 6, 7, 8, 9, 10, 11, 12, 13, 14, 15);
        const v16h b1 = __builtin_shufflevector(
            *(const v8h*)(b1p + kb), *(const v8h*)(b1p + kb + 16),
            0, 1, 2, 3, 4, 5, 6, 7, 8, 9, 10, 11, 12, 13, 14, 15);

        acc00 = __builtin_amdgcn_wmma_f32_16x16x32_f16(false, a0, false, b0, (short)0, acc00, false, false);
        acc01 = __builtin_amdgcn_wmma_f32_16x16x32_f16(false, a0, false, b1, (short)0, acc01, false, false);
        acc10 = __builtin_amdgcn_wmma_f32_16x16x32_f16(false, a1, false, b0, (short)0, acc10, false, false);
        acc11 = __builtin_amdgcn_wmma_f32_16x16x32_f16(false, a1, false, b1, (short)0, acc11, false, false);
    }

    // C/D layout: element i -> row = half*8 + i, col = lane&15
    const int nA = n0 + r, nB = n0 + 16 + r;
    const float bA = bias ? bias[nA] : 0.0f;
    const float bB = bias ? bias[nB] : 0.0f;
#pragma unroll
    for (int i = 0; i < 8; ++i) {
        const int mA = m0 + half * 8 + i;
        const int mB = mA + 16;
        C[(size_t)mA * N + nA] = acc00[i] + bA;
        C[(size_t)mA * N + nB] = acc01[i] + bB;
        C[(size_t)mB * N + nA] = acc10[i] + bA;
        C[(size_t)mB * N + nB] = acc11[i] + bB;
    }
}

// ---------------------------------------------------------------------------
// LDS-tiled transpose: out[C][R] = in[R][C]^T.  Block 256 threads, 32x32 tile.
// ---------------------------------------------------------------------------
__global__ __launch_bounds__(256)
void transpose_kernel(const float* __restrict__ in, float* __restrict__ out,
                      int R, int C) {
    __shared__ float tile[32][33];
    const int c0 = blockIdx.x * 32, r0 = blockIdx.y * 32;
    const int tx = threadIdx.x & 31, ty = threadIdx.x >> 5;   // 32 x 8
    for (int i = ty; i < 32; i += 8) {
        const int rr = r0 + i, cc = c0 + tx;
        tile[i][tx] = (rr < R && cc < C) ? in[(size_t)rr * C + cc] : 0.0f;
    }
    __syncthreads();
    for (int i = ty; i < 32; i += 8) {
        const int rr = c0 + i, cc = r0 + tx;                  // out is [C][R]
        if (rr < C && cc < R) out[(size_t)rr * R + cc] = tile[tx][i];
    }
}

// ---------------------------------------------------------------------------
// Attention scores: e[t][j] = sum_d v[j][d] * tanh(u[j][d] + w[t][d]).
// 16x16 (t,j) tile per block; rows staged in LDS (async b128 copies when
// available) with a 516-float stride so per-thread row bases hit distinct
// banks (516 % 64 == 4, 16*4 == 64).
// ---------------------------------------------------------------------------
#define SSTR 516
__global__ __launch_bounds__(256)
void score_kernel(const float* __restrict__ u, const float* __restrict__ w,
                  const float* __restrict__ v, float* __restrict__ e) {
    __shared__ float ws_[16 * SSTR];
    __shared__ float us_[16 * SSTR];
    __shared__ float vs_[16 * SSTR];

    const int t0 = blockIdx.y * 16, j0 = blockIdx.x * 16;

    // Stage 16 rows x 512 floats of w(t), u(j), v(j). 2048 16-byte chunks per
    // matrix, 256 threads -> 8 iterations, no divergence.
    for (int q = threadIdx.x; q < 16 * (DDIM / 4); q += blockDim.x) {
        const int row = q / (DDIM / 4);
        const int c4  = q % (DDIM / 4);
#if ASYNC_LDS
        __builtin_amdgcn_global_load_async_to_lds_b128(
            (g_i32x4*)(uintptr_t)(w + (size_t)(t0 + row) * DDIM + c4 * 4),
            (l_i32x4*)(uintptr_t)&ws_[row * SSTR + c4 * 4], 0, 0);
        __builtin_amdgcn_global_load_async_to_lds_b128(
            (g_i32x4*)(uintptr_t)(u + (size_t)(j0 + row) * DDIM + c4 * 4),
            (l_i32x4*)(uintptr_t)&us_[row * SSTR + c4 * 4], 0, 0);
        __builtin_amdgcn_global_load_async_to_lds_b128(
            (g_i32x4*)(uintptr_t)(v + (size_t)(j0 + row) * DDIM + c4 * 4),
            (l_i32x4*)(uintptr_t)&vs_[row * SSTR + c4 * 4], 0, 0);
#else
        float4 a = ((const float4*)(w + (size_t)(t0 + row) * DDIM))[c4];
        *((float4*)&ws_[row * SSTR + c4 * 4]) = a;
        float4 b = ((const float4*)(u + (size_t)(j0 + row) * DDIM))[c4];
        *((float4*)&us_[row * SSTR + c4 * 4]) = b;
        float4 g = ((const float4*)(v + (size_t)(j0 + row) * DDIM))[c4];
        *((float4*)&vs_[row * SSTR + c4 * 4]) = g;
#endif
    }
#if ASYNC_LDS
#if __has_builtin(__builtin_amdgcn_s_wait_asynccnt)
    __builtin_amdgcn_s_wait_asynccnt(0);
#else
    asm volatile("s_wait_asynccnt 0x0" ::: "memory");
#endif
#endif
    __syncthreads();

    const int jl = threadIdx.x & 15, tl = threadIdx.x >> 4;
    const float* __restrict__ wp = &ws_[tl * SSTR];
    const float* __restrict__ up = &us_[jl * SSTR];
    const float* __restrict__ vp = &vs_[jl * SSTR];

    float acc = 0.0f;
#pragma unroll 4
    for (int d = 0; d < DDIM; ++d)
        acc += vp[d] * tanhf(up[d] + wp[d]);

    e[(size_t)(t0 + tl) * LSEQ + (j0 + jl)] = acc;
}

// ---------------------------------------------------------------------------
// Row softmax in place over e [L][L]; one block per row.
// ---------------------------------------------------------------------------
__global__ __launch_bounds__(256)
void softmax_kernel(float* __restrict__ e) {
    __shared__ float red[256];
    float* row = e + (size_t)blockIdx.x * LSEQ;

    float m = -INFINITY;
    for (int j = threadIdx.x; j < LSEQ; j += 256) m = fmaxf(m, row[j]);
    red[threadIdx.x] = m; __syncthreads();
    for (int s = 128; s > 0; s >>= 1) {
        if (threadIdx.x < s)
            red[threadIdx.x] = fmaxf(red[threadIdx.x], red[threadIdx.x + s]);
        __syncthreads();
    }
    m = red[0]; __syncthreads();

    float sum = 0.0f;
    for (int j = threadIdx.x; j < LSEQ; j += 256) {
        float x = __expf(row[j] - m);
        row[j] = x;
        sum += x;
    }
    red[threadIdx.x] = sum; __syncthreads();
    for (int s = 128; s > 0; s >>= 1) {
        if (threadIdx.x < s) red[threadIdx.x] += red[threadIdx.x + s];
        __syncthreads();
    }
    const float inv = 1.0f / red[0];
    for (int j = threadIdx.x; j < LSEQ; j += 256) row[j] *= inv;
}

// ---------------------------------------------------------------------------
// x16 = f16(concat(v, c)) -> [L][2D] directly in f16 for the gi GEMMs.
// ---------------------------------------------------------------------------
__global__ __launch_bounds__(256)
void concat_f16_kernel(const float* __restrict__ v, const float* __restrict__ c,
                       _Float16* __restrict__ x16) {
    const int idx = blockIdx.x * blockDim.x + threadIdx.x;
    if (idx >= LSEQ * 2 * DDIM) return;
    const int t = idx / (2 * DDIM);
    const int col = idx % (2 * DDIM);
    const float val = (col < DDIM) ? v[(size_t)t * DDIM + col]
                                   : c[(size_t)t * DDIM + (col - DDIM)];
    x16[idx] = (_Float16)val;
}

// ---------------------------------------------------------------------------
// Bidirectional GRU scan. grid.x = 2 (0 = forward, 1 = backward); 512 threads.
// gi[t][3H] already contains x@wih^T + bih (f32). whhT is [H][3H] so per-k
// gate loads are coalesced and whhT (3MB/dir) stays hot in L2. h in LDS.
// ---------------------------------------------------------------------------
__global__ __launch_bounds__(512)
void gru_kernel(const float* __restrict__ gif, const float* __restrict__ gib,
                const float* __restrict__ whhfT, const float* __restrict__ whhbT,
                const float* __restrict__ bhhf, const float* __restrict__ bhhb,
                float* __restrict__ out) {
    const int rev = blockIdx.x;
    const float* __restrict__ gi   = rev ? gib   : gif;
    const float* __restrict__ whhT = rev ? whhbT : whhfT;
    const float* __restrict__ bhh  = rev ? bhhb  : bhhf;

    __shared__ float hs[HDIM];
    const int i = threadIdx.x;                       // one hidden unit each
    hs[i] = 0.0f;
    const float br = bhh[i], bz = bhh[HDIM + i], bn = bhh[2 * HDIM + i];
    __syncthreads();

    for (int s = 0; s < LSEQ; ++s) {
        const int t = rev ? (LSEQ - 1 - s) : s;
        const float* __restrict__ gp = gi + (size_t)t * 3 * HDIM;
        const int tn = rev ? (t - 1) : (t + 1);
        if (tn >= 0 && tn < LSEQ)
            __builtin_prefetch(gi + (size_t)tn * 3 * HDIM, 0, 0);

        const float hold = hs[i];
        float ar = br, az = bz, an = bn;
#pragma unroll 4
        for (int k = 0; k < HDIM; ++k) {
            const float hk = hs[k];                  // LDS broadcast
            const float* __restrict__ wr = whhT + (size_t)k * 3 * HDIM;
            ar += hk * wr[i];
            az += hk * wr[HDIM + i];
            an += hk * wr[2 * HDIM + i];
        }
        const float r = 1.0f / (1.0f + __expf(-(gp[i] + ar)));
        const float z = 1.0f / (1.0f + __expf(-(gp[HDIM + i] + az)));
        const float n = tanhf(gp[2 * HDIM + i] + r * an);
        const float hn = (1.0f - z) * n + z * hold;

        __syncthreads();           // all reads of hs done
        hs[i] = hn;
        __syncthreads();           // new h visible to everyone
        out[(size_t)t * 2 * HDIM + rev * HDIM + i] = hn;
    }
}

// ---------------------------------------------------------------------------
// Host-side orchestration (all on `stream`, graph-capture safe).
// ---------------------------------------------------------------------------
extern "C" void kernel_launch(void* const* d_in, const int* in_sizes, int n_in,
                              void* d_out, int out_size, void* d_ws, size_t ws_size,
                              hipStream_t stream) {
    (void)in_sizes; (void)n_in; (void)out_size; (void)ws_size;

    const float* v     = (const float*)d_in[0];
    const float* w1    = (const float*)d_in[1];
    const float* b1    = (const float*)d_in[2];
    const float* w2    = (const float*)d_in[3];
    const float* b2    = (const float*)d_in[4];
    const float* wih_f = (const float*)d_in[5];
    const float* whh_f = (const float*)d_in[6];
    const float* bih_f = (const float*)d_in[7];
    const float* bhh_f = (const float*)d_in[8];
    const float* wih_b = (const float*)d_in[9];
    const float* whh_b = (const float*)d_in[10];
    const float* bih_b = (const float*)d_in[11];
    const float* bhh_b = (const float*)d_in[12];
    float* out = (float*)d_out;

    const int LD  = LSEQ * DDIM;          // 262144
    const int LL  = LSEQ * LSEQ;          // 262144
    const int L3H = LSEQ * 3 * HDIM;      // 786432
    const int H3H = HDIM * 3 * HDIM;      // 786432
    const int WIH = 3 * HDIM * 2 * DDIM;  // 1572864

    // ---- workspace layout: f32 region then f16 region (~28 MB total) ----
    float* ws    = (float*)d_ws;
    float* u     = ws;                 // [L][D]
    float* w     = u     + LD;         // [L][D]
    float* e     = w     + LD;         // [L][L] (softmax in place)
    float* vT    = e     + LL;         // [D][L]
    float* c     = vT    + LD;         // [L][D]
    float* gif   = c     + LD;         // [L][3H]
    float* gib   = gif   + L3H;        // [L][3H]
    float* whhfT = gib   + L3H;        // [H][3H]
    float* whhbT = whhfT + H3H;        // [H][3H]

    _Float16* v16    = (_Float16*)(whhbT + H3H);   // [L][D]
    _Float16* w1h    = v16    + LD;                // [D][D]
    _Float16* w2h    = w1h    + LD;                // [D][D]
    _Float16* vT16   = w2h    + LD;                // [D][L]
    _Float16* a16    = vT16   + LD;                // [L][L]
    _Float16* x16    = a16    + LL;                // [L][2D]
    _Float16* wihf16 = x16    + LSEQ * 2 * DDIM;   // [3H][2D]
    _Float16* wihb16 = wihf16 + WIH;               // [3H][2D]

    // 1) transposes: vT = v^T ; whh{f,b}T = whh^T  ([3H][H] -> [H][3H])
    transpose_kernel<<<dim3(DDIM / 32, LSEQ / 32), 256, 0, stream>>>(v, vT, LSEQ, DDIM);
    transpose_kernel<<<dim3(HDIM / 32, (3 * HDIM) / 32), 256, 0, stream>>>(whh_f, whhfT, 3 * HDIM, HDIM);
    transpose_kernel<<<dim3(HDIM / 32, (3 * HDIM) / 32), 256, 0, stream>>>(whh_b, whhbT, 3 * HDIM, HDIM);

    // 2) one-time f16 conversions of static GEMM operands
    f32_to_f16_kernel<<<(LD / 4 + 255) / 256, 256, 0, stream>>>(v, v16, LD);
    f32_to_f16_kernel<<<(LD / 4 + 255) / 256, 256, 0, stream>>>(w1, w1h, LD);
    f32_to_f16_kernel<<<(LD / 4 + 255) / 256, 256, 0, stream>>>(w2, w2h, LD);
    f32_to_f16_kernel<<<(LD / 4 + 255) / 256, 256, 0, stream>>>(vT, vT16, LD);
    f32_to_f16_kernel<<<(WIH / 4 + 255) / 256, 256, 0, stream>>>(wih_f, wihf16, WIH);
    f32_to_f16_kernel<<<(WIH / 4 + 255) / 256, 256, 0, stream>>>(wih_b, wihb16, WIH);

    // 3) u = v@w1^T + b1 ; w = v@w2^T + b2   (WMMA, 128x64 block tiles)
    gemm_wmma_f16_kernel<<<dim3(DDIM / 64, LSEQ / 128), 256, 0, stream>>>(v16, w1h, b1, u, LSEQ, DDIM, DDIM);
    gemm_wmma_f16_kernel<<<dim3(DDIM / 64, LSEQ / 128), 256, 0, stream>>>(v16, w2h, b2, w, LSEQ, DDIM, DDIM);

    // 4) e[t][j] = sum_d v[j][d] * tanh(u[j][d] + w[t][d])
    score_kernel<<<dim3(LSEQ / 16, LSEQ / 16), 256, 0, stream>>>(u, w, v, e);

    // 5) a = softmax(e) rows, in place; then a -> f16
    softmax_kernel<<<LSEQ, 256, 0, stream>>>(e);
    f32_to_f16_kernel<<<(LL / 4 + 255) / 256, 256, 0, stream>>>(e, a16, LL);

    // 6) c = a @ v   (W = v^T so W[n][k] = v[k][n])  (WMMA)
    gemm_wmma_f16_kernel<<<dim3(DDIM / 64, LSEQ / 128), 256, 0, stream>>>(a16, vT16, nullptr, c, LSEQ, DDIM, LSEQ);

    // 7) x16 = f16([v | c])
    concat_f16_kernel<<<(LSEQ * 2 * DDIM + 255) / 256, 256, 0, stream>>>(v, c, x16);

    // 8) gi = x @ wih^T + bih for both directions  (WMMA)
    gemm_wmma_f16_kernel<<<dim3((3 * HDIM) / 64, LSEQ / 128), 256, 0, stream>>>(x16, wihf16, bih_f, gif, LSEQ, 3 * HDIM, 2 * DDIM);
    gemm_wmma_f16_kernel<<<dim3((3 * HDIM) / 64, LSEQ / 128), 256, 0, stream>>>(x16, wihb16, bih_b, gib, LSEQ, 3 * HDIM, 2 * DDIM);

    // 9) bidirectional GRU scan (fwd + bwd concurrently)
    gru_kernel<<<2, 512, 0, stream>>>(gif, gib, whhfT, whhbT, bhh_f, bhh_b, out);
}
